// SelfAttentionV2_51599737094690
// MI455X (gfx1250) — matrix operand order repaired and checked by reading, not compiled, not run on hardware
//
#include <hip/hip_runtime.h>

// Problem constants (from the reference)
#define HID 1024
#define SEQ 2048
#define NB  4
#define MROWS (NB * SEQ)      // 8192
#define N3   (3 * HID)        // 3072

typedef __attribute__((ext_vector_type(16))) __bf16 v16bf;
typedef __attribute__((ext_vector_type(8)))  __bf16 v8bf;
typedef __attribute__((ext_vector_type(8)))  float  v8f;
typedef __attribute__((ext_vector_type(4)))  float  v4f;

// ---------------------------------------------------------------------------
// A-matrix fragment (16x32 bf16), source row-major along K.
// Lane layout (contiguous K, same order as B): lane L supplies row M=L%16, g = L/16:
//   halfs a[0..7]  hold K = kk0 + g*16 + j
//   halfs a[8..15] hold K = kk0 + g*16 + 8 + j
// ---------------------------------------------------------------------------
__device__ __forceinline__ v16bf load_a_frag(const __bf16* __restrict__ row,
                                             int kk0, int g) {
    v8bf lo = *(const v8bf*)(row + kk0 + g * 16);
    v8bf hi = *(const v8bf*)(row + kk0 + g * 16 + 8);
    v16bf a;
#pragma unroll
    for (int j = 0; j < 8; ++j) {
        a[j]     = lo[j];
        a[8 + j] = hi[j];
    }
    return a;
}

// ---------------------------------------------------------------------------
// Async-stage one 64-row x 32-col bf16 B-tile (4 KB) into LDS.
// Wave-cooperative: 8 x global_load_async_to_lds_b128, 16 B per lane each.
// LDS image: 64 rows x 64 bytes, row-major.  Tracked by ASYNCcnt.
// ---------------------------------------------------------------------------
__device__ __forceinline__ void stage_b_tile(const __bf16* tile0,   // &B[row0][kk0]
                                             long stride_elems,     // row stride
                                             unsigned lds_buf,      // LDS byte offset
                                             int lane) {
#pragma unroll
    for (int a = 0; a < 8; ++a) {
        const int e   = a * 512 + lane * 16;   // byte offset within 4 KB tile
        const int row = e >> 6;                // 64 B per row
        const int col = e & 63;
        const unsigned long long g = (unsigned long long)
            ((const char*)(tile0 + (long)row * stride_elems) + col);
        const unsigned l = lds_buf + (unsigned)e;
        asm volatile("global_load_async_to_lds_b128 %0, %1, off"
                     :: "v"(l), "v"(g) : "memory");
    }
}

// B-fragment (32x16) read from a staged LDS tile: lane supplies column
// N = l16 (= tile row t*16+l16), halfs b[j] hold K = g*16 + j  -> 32 B ds read.
__device__ __forceinline__ v16bf load_b_frag_lds(const __bf16* sb, int t,
                                                 int l16, int g) {
    return *(const v16bf*)(sb + (t * 16 + l16) * 32 + g * 16);
}

// ===========================================================================
// Kernel 0: f32 -> bf16 conversion (x and W), 8 elements/thread.
// ===========================================================================
__global__ void __launch_bounds__(256)
cvt_bf16_kernel(const float* __restrict__ in, __bf16* __restrict__ out) {
    const long i = ((long)blockIdx.x * 256 + threadIdx.x) * 8;
    v4f a = *(const v4f*)(in + i);
    v4f b = *(const v4f*)(in + i + 4);
    v8bf o;
#pragma unroll
    for (int j = 0; j < 4; ++j) {
        o[j]     = (__bf16)a[j];
        o[4 + j] = (__bf16)b[j];
    }
    *(v8bf*)(out + i) = o;
}

// ---------------------------------------------------------------------------
// Shared 64x64 wave-tile GEMM inner pipeline (bf16 x bf16 -> f32), with
// double-buffered async LDS staging of the B operand.
//   arow[t] : A row pointers (row-major along K)
//   btile0  : &B[tile_row0][0]   (row-major along K, stride bstride)
// ---------------------------------------------------------------------------
#define GEMM_PIPELINE(KD, arow, btile0, bstride, acc, sB, l16, g, lane)        \
    do {                                                                       \
        const unsigned lb0 = (unsigned)(unsigned long long)(&sB[0][0]);        \
        const unsigned lb1 = (unsigned)(unsigned long long)(&sB[1][0]);        \
        stage_b_tile(btile0, bstride, lb0, lane);                              \
        int cur = 0;                                                           \
        for (int kk0 = 0; kk0 < (KD); kk0 += 32) {                             \
            if (kk0 + 32 < (KD)) {                                             \
                stage_b_tile((btile0) + kk0 + 32, bstride,                     \
                             cur ? lb0 : lb1, lane);                           \
                asm volatile("s_wait_asynccnt 0x8" ::: "memory");              \
            } else {                                                           \
                asm volatile("s_wait_asynccnt 0x0" ::: "memory");              \
            }                                                                  \
            const __bf16* sb = cur ? &sB[1][0] : &sB[0][0];                    \
            v16bf af[4], bf_[4];                                               \
            _Pragma("unroll")                                                  \
            for (int t = 0; t < 4; ++t) af[t] = load_a_frag(arow[t], kk0, g);  \
            _Pragma("unroll")                                                  \
            for (int t = 0; t < 4; ++t) bf_[t] = load_b_frag_lds(sb, t, l16, g); \
            _Pragma("unroll")                                                  \
            for (int ti = 0; ti < 4; ++ti)                                     \
                _Pragma("unroll")                                              \
                for (int tj = 0; tj < 4; ++tj)                                 \
                    acc[ti][tj] = __builtin_amdgcn_wmma_f32_16x16x32_bf16(     \
                        false, af[ti], false, bf_[tj], (short)0, acc[ti][tj],  \
                        false, false);                                         \
            cur ^= 1;                                                          \
        }                                                                      \
    } while (0)

// ===========================================================================
// Kernel 1: fused QKV projection  y[m,o] = sum_h x[m,h]*W[o,h] + bias[o]
//   One wave computes a 64x64 tile.  Writes Q, K [B,S,H], Vt [B,H,S] bf16.
// ===========================================================================
__global__ void __launch_bounds__(32)
qkv_gemm_kernel(const __bf16* __restrict__ Xb,    // [M, H] bf16
                const __bf16* __restrict__ Wb,    // [3H, H] bf16
                const float*  __restrict__ bias,  // [3H]
                __bf16* __restrict__ Qb,
                __bf16* __restrict__ Kb,
                __bf16* __restrict__ Vt) {
    __shared__ __align__(16) __bf16 sB[2][64 * 32];
    const int lane = threadIdx.x & 31;
    const int l16  = lane & 15;
    const int g    = lane >> 4;

    const int m0 = blockIdx.x * 64;
    const int n0 = blockIdx.y * 64;

    const __bf16* arow[4];
#pragma unroll
    for (int t = 0; t < 4; ++t)
        arow[t] = Xb + (long)(m0 + t * 16 + l16) * HID;
    const __bf16* btile0 = Wb + (long)n0 * HID;

    v8f acc[4][4] = {};
    GEMM_PIPELINE(HID, arow, btile0, (long)HID, acc, sB, l16, g, lane);

#pragma unroll
    for (int tj = 0; tj < 4; ++tj) {
        const int   ocol = n0 + tj * 16 + l16;
        const float bv   = bias[ocol];
#pragma unroll
        for (int ti = 0; ti < 4; ++ti) {
#pragma unroll
            for (int i = 0; i < 8; ++i) {
                const int    m    = m0 + ti * 16 + g * 8 + i;
                const int    bidx = m >> 11;          // / SEQ
                const int    s    = m & (SEQ - 1);
                const __bf16 h    = (__bf16)(acc[ti][tj][i] + bv);
                if (ocol < HID) {                     // Q
                    Qb[(long)bidx * SEQ * HID + (long)s * HID + ocol] = h;
                } else if (ocol < 2 * HID) {          // K
                    Kb[(long)bidx * SEQ * HID + (long)s * HID + (ocol - HID)] = h;
                } else {                              // V, transposed [H,S]
                    Vt[(long)bidx * HID * SEQ + (long)(ocol - 2 * HID) * SEQ + s] = h;
                }
            }
        }
    }
}

// ===========================================================================
// Kernel 2: scores[b,q,k] = (1/sqrt(H)) * sum_h Q[b,q,h]*K[b,k,h]
// ===========================================================================
__global__ void __launch_bounds__(32)
scores_kernel(const __bf16* __restrict__ Qb,
              const __bf16* __restrict__ Kb,
              float* __restrict__ Sc) {
    __shared__ __align__(16) __bf16 sB[2][64 * 32];
    const int lane = threadIdx.x & 31;
    const int l16  = lane & 15;
    const int g    = lane >> 4;

    const int b  = blockIdx.z;
    const int q0 = blockIdx.x * 64;
    const int k0 = blockIdx.y * 64;

    const __bf16* arow[4];
#pragma unroll
    for (int t = 0; t < 4; ++t)
        arow[t] = Qb + (long)b * SEQ * HID + (long)(q0 + t * 16 + l16) * HID;
    const __bf16* btile0 = Kb + (long)b * SEQ * HID + (long)k0 * HID;

    v8f acc[4][4] = {};
    GEMM_PIPELINE(HID, arow, btile0, (long)HID, acc, sB, l16, g, lane);

    const float scale = 0.03125f;  // 1/sqrt(1024)
#pragma unroll
    for (int ti = 0; ti < 4; ++ti) {
        float* srow = Sc + (long)b * SEQ * SEQ
                        + (long)(q0 + ti * 16 + g * 8) * SEQ + k0 + l16;
#pragma unroll
        for (int tj = 0; tj < 4; ++tj)
#pragma unroll
            for (int i = 0; i < 8; ++i)
                srow[(long)i * SEQ + tj * 16] = acc[ti][tj][i] * scale;
    }
}

// ===========================================================================
// Kernel 3: row softmax over 2048 f32 scores -> bf16 probabilities.
// ===========================================================================
__global__ void __launch_bounds__(256)
softmax_kernel(const float* __restrict__ Sc, __bf16* __restrict__ P) {
    const int  tid = threadIdx.x;
    const long row = blockIdx.x;                   // b*SEQ + q
    const float* srow = Sc + row * (long)SEQ;
    __bf16*      prow = P  + row * (long)SEQ;

    float v[8];
    float mx = -3.402823466e38f;
#pragma unroll
    for (int i = 0; i < 8; ++i) {
        v[i] = srow[tid + i * 256];
        mx   = fmaxf(mx, v[i]);
    }

    __shared__ float red[256];
    red[tid] = mx;
    __syncthreads();
#pragma unroll
    for (int st = 128; st > 0; st >>= 1) {
        if (tid < st) red[tid] = fmaxf(red[tid], red[tid + st]);
        __syncthreads();
    }
    const float rowmax = red[0];
    __syncthreads();

    float sum = 0.0f;
#pragma unroll
    for (int i = 0; i < 8; ++i) {
        v[i] = __expf(v[i] - rowmax);
        sum += v[i];
    }
    red[tid] = sum;
    __syncthreads();
#pragma unroll
    for (int st = 128; st > 0; st >>= 1) {
        if (tid < st) red[tid] += red[tid + st];
        __syncthreads();
    }
    const float inv = 1.0f / red[0];

#pragma unroll
    for (int i = 0; i < 8; ++i) {
        prow[tid + i * 256] = (__bf16)(v[i] * inv);
    }
}

// ===========================================================================
// Kernel 4: out[b,q,h] = sum_k P[b,q,k] * V[b,k,h]
//   A = P rows, B(kk,n) = Vt[b][h][k], contiguous in k (stride SEQ rows).
// ===========================================================================
__global__ void __launch_bounds__(32)
out_gemm_kernel(const __bf16* __restrict__ P,
                const __bf16* __restrict__ Vt,
                float* __restrict__ out) {
    __shared__ __align__(16) __bf16 sB[2][64 * 32];
    const int lane = threadIdx.x & 31;
    const int l16  = lane & 15;
    const int g    = lane >> 4;

    const int b  = blockIdx.z;
    const int q0 = blockIdx.x * 64;
    const int h0 = blockIdx.y * 64;

    const __bf16* arow[4];
#pragma unroll
    for (int t = 0; t < 4; ++t)
        arow[t] = P + (long)b * SEQ * SEQ + (long)(q0 + t * 16 + l16) * SEQ;
    const __bf16* btile0 = Vt + (long)b * HID * SEQ + (long)h0 * SEQ;

    v8f acc[4][4] = {};
    GEMM_PIPELINE(SEQ, arow, btile0, (long)SEQ, acc, sB, l16, g, lane);

#pragma unroll
    for (int ti = 0; ti < 4; ++ti) {
        float* orow = out + (long)b * SEQ * HID
                          + (long)(q0 + ti * 16 + g * 8) * HID + h0 + l16;
#pragma unroll
        for (int tj = 0; tj < 4; ++tj)
#pragma unroll
            for (int i = 0; i < 8; ++i)
                orow[(long)i * HID + tj * 16] = acc[ti][tj][i];
    }
}

// ===========================================================================
// Workspace layout (bytes), requires ws_size >= 134 MiB:
//   [0,   16M)  Xb bf16          -- later overwritten by P bf16 [0, 32M)
//   [16M, 22M)  Wb bf16
//   [22M, 38M)  Q  bf16          -- dead after scores; head reused by P
//   [38M, 54M)  K  bf16
//   [54M, 70M)  Vt bf16
//   [70M, 134M) scores f32
// ===========================================================================
extern "C" void kernel_launch(void* const* d_in, const int* in_sizes, int n_in,
                              void* d_out, int out_size, void* d_ws, size_t ws_size,
                              hipStream_t stream) {
    const float* x    = (const float*)d_in[0];   // [B,S,H]
    const float* W    = (const float*)d_in[1];   // [3H,H]
    const float* bias = (const float*)d_in[2];   // [3H]
    float* out        = (float*)d_out;           // [B,S,H]

    char* ws = (char*)d_ws;
    const size_t MB = 1024u * 1024u;
    __bf16* Xb = (__bf16*)(ws + 0);
    __bf16* Wb = (__bf16*)(ws + 16 * MB);
    __bf16* Qb = (__bf16*)(ws + 22 * MB);
    __bf16* Kb = (__bf16*)(ws + 38 * MB);
    __bf16* Vt = (__bf16*)(ws + 54 * MB);
    float*  Sc = (float*) (ws + 70 * MB);
    __bf16* P  = (__bf16*)(ws + 0);              // overwrites dead Xb/Wb/Q

    // 0) f32 -> bf16 for activations and weights.
    cvt_bf16_kernel<<<dim3((MROWS * HID) / 8 / 256), 256, 0, stream>>>(x, Xb);
    cvt_bf16_kernel<<<dim3((N3 * HID) / 8 / 256), 256, 0, stream>>>(W, Wb);

    // 1) QKV projection: 64x64 tile per wave, async-staged B.
    qkv_gemm_kernel<<<dim3(MROWS / 64, N3 / 64, 1), 32, 0, stream>>>(
        Xb, Wb, bias, Qb, Kb, Vt);

    // 2) scores = scale * Q K^T.
    scores_kernel<<<dim3(SEQ / 64, SEQ / 64, NB), 32, 0, stream>>>(Qb, Kb, Sc);

    // 3) row softmax -> bf16 P.
    softmax_kernel<<<dim3(NB * SEQ, 1, 1), 256, 0, stream>>>(Sc, P);

    // 4) out = P V.
    out_gemm_kernel<<<dim3(SEQ / 64, HID / 64, NB), 32, 0, stream>>>(P, Vt, out);
}
